// _DotProductionAttention_25915832664125
// MI455X (gfx1250) — matrix operand order repaired and checked by reading, not compiled, not run on hardware
//
#include <hip/hip_runtime.h>

typedef __attribute__((ext_vector_type(16))) _Float16 v16h;
typedef __attribute__((ext_vector_type(8)))  _Float16 v8h;
typedef __attribute__((ext_vector_type(8)))  float    v8f;
typedef __attribute__((ext_vector_type(4)))  float    v4f;

constexpr int BATCH = 16;
constexpr int SEQ   = 2048;
constexpr int DIM   = 128;
constexpr int QT    = 128;          // query rows per block (8 waves x 16)
constexpr int KT    = 64;           // kv rows per tile
constexpr int NTILE = SEQ / KT;
constexpr int KSTR  = 144;          // halves per K row in LDS (row stride 288B, 32B aligned)
constexpr int VTSTR = 80;           // halves per Vt row in LDS (row stride 160B, 32B aligned)
constexpr int PSTRH = 80;           // halves per P row in LDS (row stride 160B, 16B aligned)

__global__ __launch_bounds__(256, 1)
void fattn_wmma_kernel(const float* __restrict__ Q, const float* __restrict__ K,
                       const float* __restrict__ V, float* __restrict__ Out) {
  __shared__ alignas(32) _Float16 Ks[2][KT * KSTR];     // K tile, row-major  [key][d], double buffered
  __shared__ alignas(32) _Float16 Vt[2][DIM * VTSTR];   // V tile, transposed [d][key], double buffered
  __shared__ alignas(32) _Float16 Ph[8 * 16 * PSTRH];   // per-wave P transpose scratch (f16)

  const int tid  = threadIdx.x;
  const int wave = tid >> 5;
  const int lane = tid & 31;
  const int lh   = lane >> 4;      // lane half (0: lanes 0-15, 1: lanes 16-31)
  const int ln   = lane & 15;

  const int b     = blockIdx.x >> 4;   // batch
  const int qt    = blockIdx.x & 15;   // query tile
  const int qrow0 = qt * QT + wave * 16;

  const float scale = 0.088388347648318447f; // 1/sqrt(128), folded into Q

  const float* gK = K + (size_t)b * SEQ * DIM;
  const float* gV = V + (size_t)b * SEQ * DIM;
  const int r0 = tid >> 5;           // staging row base 0..7
  const int c4 = (tid & 31) * 4;     // staging column 0,4,...,124

  // ---- Q fragments (A-layout), softmax scale pre-folded
  v16h qf[4];
  {
    const float* qrow = Q + ((size_t)b * SEQ + qrow0 + ln) * DIM;
    #pragma unroll
    for (int c = 0; c < 4; ++c) {
      const int d0 = c * 32 + lh * 16;
      #pragma unroll
      for (int i = 0; i < 8; ++i) {
        qf[c][i]     = (_Float16)(qrow[d0 + i] * scale);
        qf[c][i + 8] = (_Float16)(qrow[d0 + 8 + i] * scale);
      }
    }
  }

  // ---- accumulators
  v8f of[8];
  #pragma unroll
  for (int n = 0; n < 8; ++n) of[n] = {};
  float mrow[8], lrow[8];
  #pragma unroll
  for (int j = 0; j < 8; ++j) { mrow[j] = -1e30f; lrow[j] = 0.0f; }

  // ---- pipelined staging state (held in VGPRs across compute)
  v4f kreg[8], vreg[8];

  auto loadK = [&](int kt0) {
    #pragma unroll
    for (int rr = 0; rr < 8; ++rr)
      kreg[rr] = *(const v4f*)(gK + (size_t)(kt0 + rr * 8 + r0) * DIM + c4);
  };
  auto loadV = [&](int kt0) {
    #pragma unroll
    for (int rr = 0; rr < 8; ++rr)
      vreg[rr] = *(const v4f*)(gV + (size_t)(kt0 + rr * 8 + r0) * DIM + c4);
  };
  auto storeK = [&](int buf) {
    _Float16* kb = Ks[buf];
    #pragma unroll
    for (int rr = 0; rr < 8; ++rr) {
      _Float16* kd = &kb[(rr * 8 + r0) * KSTR + c4];
      kd[0] = (_Float16)kreg[rr].x; kd[1] = (_Float16)kreg[rr].y;
      kd[2] = (_Float16)kreg[rr].z; kd[3] = (_Float16)kreg[rr].w;
    }
  };
  auto storeV = [&](int buf) {
    _Float16* vb = Vt[buf];
    #pragma unroll
    for (int rr = 0; rr < 8; ++rr) {
      const int row = rr * 8 + r0;
      vb[(c4 + 0) * VTSTR + row] = (_Float16)vreg[rr].x;
      vb[(c4 + 1) * VTSTR + row] = (_Float16)vreg[rr].y;
      vb[(c4 + 2) * VTSTR + row] = (_Float16)vreg[rr].z;
      vb[(c4 + 3) * VTSTR + row] = (_Float16)vreg[rr].w;
    }
  };

  // ---- prologue: stage tile 0 into buffer 0
  loadK(0); loadV(0); storeK(0); storeV(0);

  for (int it = 0; it < NTILE; ++it) {
    const int p = it & 1;
    __syncthreads();  // buf p staged by all waves; buf 1-p free (compute it-1 done everywhere)

    const bool more = (it + 1 < NTILE);
    if (more) loadK((it + 1) * KT);   // issue next K loads; latency hidden by QK^T

    // ---- S = Q K^T (scale pre-folded): 4 key N-tiles x 4 d-chunks
    const _Float16* ksb = Ks[p];
    v8f s[4];
    #pragma unroll
    for (int t = 0; t < 4; ++t) {
      v8f acc = {};
      #pragma unroll
      for (int c = 0; c < 4; ++c) {
        const v16h bf = *(const v16h*)&ksb[(t * 16 + ln) * KSTR + c * 32 + lh * 16];
        acc = __builtin_amdgcn_wmma_f32_16x16x32_f16(false, qf[c], false, bf,
                                                     (short)0, acc, false, false);
      }
      s[t] = acc;
    }

    // ---- online softmax (rows M = j + 8*lh; butterfly reduce across 16-lane halves)
    float alpha[8];
    #pragma unroll
    for (int j = 0; j < 8; ++j) {
      float tm = fmaxf(fmaxf(s[0][j], s[1][j]), fmaxf(s[2][j], s[3][j]));
      #pragma unroll
      for (int msk = 1; msk < 16; msk <<= 1) tm = fmaxf(tm, __shfl_xor(tm, msk, 32));
      const float mn = fmaxf(mrow[j], tm);
      alpha[j] = __expf(mrow[j] - mn);
      mrow[j]  = mn;
      float rs = 0.0f;
      #pragma unroll
      for (int t = 0; t < 4; ++t) { const float pe = __expf(s[t][j] - mn); s[t][j] = pe; rs += pe; }
      #pragma unroll
      for (int msk = 1; msk < 16; msk <<= 1) rs += __shfl_xor(rs, msk, 32);
      lrow[j] = lrow[j] * alpha[j] + rs;
    }

    // next V loads + retire K staging regs into the other buffer
    if (more) { loadV((it + 1) * KT); storeK(1 - p); }

    #pragma unroll
    for (int n = 0; n < 8; ++n)
      #pragma unroll
      for (int j = 0; j < 8; ++j) of[n][j] *= alpha[j];

    // ---- transpose P (C-layout -> A-layout) via per-wave f16 LDS scratch (same-wave DS in-order)
    _Float16* pw = &Ph[wave * 16 * PSTRH];
    #pragma unroll
    for (int t = 0; t < 4; ++t)
      #pragma unroll
      for (int j = 0; j < 8; ++j)
        pw[(j + 8 * lh) * PSTRH + t * 16 + ln] = (_Float16)s[t][j];

    v16h pf[2];
    #pragma unroll
    for (int cc = 0; cc < 2; ++cc) {
      const v8h lo = *(const v8h*)&pw[ln * PSTRH + cc * 32 + lh * 16];
      const v8h hi = *(const v8h*)&pw[ln * PSTRH + cc * 32 + lh * 16 + 8];
      #pragma unroll
      for (int i = 0; i < 8; ++i) { pf[cc][i] = lo[i]; pf[cc][i + 8] = hi[i]; }
    }

    // ---- O += P V : 8 d N-tiles x 2 key chunks
    const _Float16* vtb = Vt[p];
    #pragma unroll
    for (int n = 0; n < 8; ++n) {
      #pragma unroll
      for (int cc = 0; cc < 2; ++cc) {
        const v16h vf = *(const v16h*)&vtb[(n * 16 + ln) * VTSTR + cc * 32 + lh * 16];
        of[n] = __builtin_amdgcn_wmma_f32_16x16x32_f16(false, pf[cc], false, vf,
                                                       (short)0, of[n], false, false);
      }
    }

    // retire V staging regs (their load latency was covered by softmax + PV)
    if (more) storeV(1 - p);

    if (it + 2 < NTILE) {  // pull tile it+2 toward L2 (global_prefetch_b8)
      __builtin_prefetch(gK + (size_t)((it + 2) * KT + r0 * 8) * DIM + c4, 0, 1);
      __builtin_prefetch(gV + (size_t)((it + 2) * KT + r0 * 8) * DIM + c4, 0, 1);
    }
  }

  // ---- epilogue: normalize and store (rows M = j + 8*lh, cols d = n*16 + ln)
  float* orow = Out + ((size_t)b * SEQ + qrow0) * DIM;
  #pragma unroll
  for (int j = 0; j < 8; ++j) {
    const float inv = 1.0f / lrow[j];
    #pragma unroll
    for (int n = 0; n < 8; ++n)
      orow[(j + 8 * lh) * DIM + n * 16 + ln] = of[n][j] * inv;
  }
}

extern "C" void kernel_launch(void* const* d_in, const int* in_sizes, int n_in,
                              void* d_out, int out_size, void* d_ws, size_t ws_size,
                              hipStream_t stream) {
  (void)in_sizes; (void)n_in; (void)out_size; (void)d_ws; (void)ws_size;
  const float* Q = (const float*)d_in[0];
  const float* K = (const float*)d_in[1];
  const float* V = (const float*)d_in[2];
  float* O = (float*)d_out;
  dim3 grid(BATCH * (SEQ / QT));   // 16 batches x 16 query tiles = 256 blocks
  fattn_wmma_kernel<<<grid, dim3(256), 0, stream>>>(Q, K, V, O);
}
